// OnnxMultiHeadAttention_81698867904536
// MI455X (gfx1250) — matrix operand order repaired and checked by reading, not compiled, not run on hardware
//
#include <hip/hip_runtime.h>
#include <math.h>

typedef __attribute__((ext_vector_type(16))) _Float16 v16h;
typedef __attribute__((ext_vector_type(8)))  _Float16 v8h;
typedef __attribute__((ext_vector_type(4)))  _Float16 v4h;
typedef __attribute__((ext_vector_type(8)))  float    v8f;

#define S_LEN 2048
#define DM    1024
#define NH    16
#define HD    64

// ds_swizzle_b32, group-of-32 mode: offset = xor<<10 | or<<5 | and(0x1f).
// Constant-imm lane XOR swap: one DS op, no index math, no bounds check.
#define SWZ_F32(x, imm) \
  __int_as_float(__builtin_amdgcn_ds_swizzle(__float_as_int(x), (imm)))
#define XOR1(x)  SWZ_F32((x), 0x041f)
#define XOR2(x)  SWZ_F32((x), 0x081f)
#define XOR4(x)  SWZ_F32((x), 0x101f)
#define XOR8(x)  SWZ_F32((x), 0x201f)

__device__ __forceinline__ v8f wmma16x16x32(v16h a, v16h b, v8f c) {
  // (neg_a, A, neg_b, B, c_mod, C, reuse_a, reuse_b)
  return __builtin_amdgcn_wmma_f32_16x16x32_f16(false, a, false, b, (short)0, c,
                                                false, false);
}

// A-operand fragment (16x32, MxK), row-major [M][K] source (global or LDS).
// ISA 7.12.2: lane L holds M=L%16; K halves {kh..kh+7} and {kh+16..kh+23},
// kh = (L/16)*8.
__device__ __forceinline__ v16h load_frag_a(const _Float16* base, int ld, int lane) {
  const _Float16* p = base + (lane & 15) * ld + ((lane >> 4) << 3);
  v8h lo = *(const v8h*)p;
  v8h hi = *(const v8h*)(p + 16);
  v16h r;
#pragma unroll
  for (int i = 0; i < 8; ++i) { r[i] = lo[i]; r[i + 8] = hi[i]; }
  return r;
}

// B-operand fragment (32x16, KxN) from row-major [N][K] source:
// lane L holds N=L%16; K halves {kh..kh+7} and {kh+16..kh+23}, kh = (L/16)*8.
__device__ __forceinline__ v16h load_frag_b(const _Float16* base, int ld, int lane) {
  const _Float16* p = base + (lane & 15) * ld + ((lane >> 4) << 3);
  v8h lo = *(const v8h*)p;
  v8h hi = *(const v8h*)(p + 16);
  v16h r;
#pragma unroll
  for (int i = 0; i < 8; ++i) { r[i] = lo[i]; r[i + 8] = hi[i]; }
  return r;
}

// ---------------------------------------------------------------- f32 -> f16
__global__ void cvt_f32_f16(const float* __restrict__ in,
                            _Float16* __restrict__ out, int n) {
  int i = (blockIdx.x * blockDim.x + threadIdx.x) * 4;
  if (i < n) {
    float4 v = *(const float4*)(in + i);
    v4h o = { (_Float16)v.x, (_Float16)v.y, (_Float16)v.z, (_Float16)v.w };
    *(v4h*)(out + i) = o;
  }
}

// ------------------------------------------------- QKV projection (+ RoPE)
// C[m,n] = sum_k X[m,k] * W[n,k];  out scattered to [B,H,S,64] f16.
// 128x128 block tile, k-tile 64 (2 WMMA k-steps per LDS stage).
__global__ __launch_bounds__(256) void qkv_gemm_rope(
    const _Float16* __restrict__ Xh, const _Float16* __restrict__ Wh,
    const float* __restrict__ cosT, const float* __restrict__ sinT,
    _Float16* __restrict__ outH, int do_rope) {
  __shared__ __attribute__((aligned(16))) _Float16 lA[128 * 72];
  __shared__ __attribute__((aligned(16))) _Float16 lB[128 * 72];
  const int tid  = threadIdx.x;
  const int lane = tid & 31, wv = tid >> 5;
  const int wm = (wv & 3) * 32, wn = (wv >> 2) * 64;
  const int bM = blockIdx.x * 128, bN = blockIdx.y * 128;
  const int lr = tid >> 1, lc = (tid & 1) << 5;  // 64-wide k-slab per row

  v8f acc[2][4] = {};

  for (int k0 = 0; k0 < DM; k0 += 64) {
    __syncthreads();
    const _Float16* ga = Xh + (size_t)(bM + lr) * DM + k0 + lc;
    const _Float16* gb = Wh + (size_t)(bN + lr) * DM + k0 + lc;
    uint4* sa = (uint4*)&lA[lr * 72 + lc];
    uint4* sb = (uint4*)&lB[lr * 72 + lc];
#pragma unroll
    for (int j = 0; j < 4; ++j) sa[j] = ((const uint4*)ga)[j];
#pragma unroll
    for (int j = 0; j < 4; ++j) sb[j] = ((const uint4*)gb)[j];
    __builtin_prefetch(ga + 64, 0, 0);  // next k-panel -> global_prefetch_b8
    __builtin_prefetch(gb + 64, 0, 0);
    __syncthreads();

#pragma unroll
    for (int kt = 0; kt < 2; ++kt) {
      v16h af[2], bf[4];
#pragma unroll
      for (int mt = 0; mt < 2; ++mt)
        af[mt] = load_frag_a(&lA[(wm + mt * 16) * 72 + kt * 32], 72, lane);
#pragma unroll
      for (int nt = 0; nt < 4; ++nt)
        bf[nt] = load_frag_b(&lB[(wn + nt * 16) * 72 + kt * 32], 72, lane);
#pragma unroll
      for (int mt = 0; mt < 2; ++mt)
#pragma unroll
        for (int nt = 0; nt < 4; ++nt)
          acc[mt][nt] = wmma16x16x32(af[mt], bf[nt], acc[mt][nt]);
    }
  }

#pragma unroll
  for (int mt = 0; mt < 2; ++mt)
#pragma unroll
    for (int nt = 0; nt < 4; ++nt) {
      const int n = bN + wn + nt * 16 + (lane & 15);
      const int h = n >> 6, d = n & 63;
      const float sgn = (d & 1) ? 1.0f : -1.0f;  // branchless RoPE sign
#pragma unroll
      for (int r = 0; r < 8; ++r) {
        const int m = bM + wm + mt * 16 + ((lane >> 4) << 3) + r;
        const int s = m & (S_LEN - 1), b = m >> 11;
        const float v = acc[mt][nt][r];
        float o;
        if (do_rope) {
          // pair (2i,2i+1) of the head dim sits in adjacent lanes
          const float pr = XOR1(v);
          const int fi = d >> 1;
          const float cf = cosT[s * (HD / 2) + fi];
          const float sf = sinT[s * (HD / 2) + fi];
          o = fmaf(pr * sgn, sf, v * cf);  // even: v*c - p*s ; odd: v*c + p*s
        } else {
          o = v;
        }
        outH[((size_t)((b * NH + h) * S_LEN + s) << 6) + d] = (_Float16)o;
      }
    }
}

// -------------------------------------------------------- flash attention
// grid.x = B*NH, grid.y = S/128; 8 waves, each owns 16 query rows.
__global__ __launch_bounds__(256) void flash_attn(
    const _Float16* __restrict__ Qh, const _Float16* __restrict__ Kh,
    const _Float16* __restrict__ Vh, _Float16* __restrict__ Ch) {
  __shared__ __attribute__((aligned(16))) _Float16 sK[32 * 72];    // keys x d
  __shared__ __attribute__((aligned(16))) _Float16 sVT[64 * 40];   // d x keys
  __shared__ __attribute__((aligned(16))) _Float16 sP[8 * 16 * 40];// per wave

  const int bh = blockIdx.x;
  const int b = bh >> 4, h = bh & 15;
  const int qblk = blockIdx.y * 128;
  const int tid = threadIdx.x, lane = tid & 31, wv = tid >> 5;
  const int q0 = qblk + wv * 16;

  const _Float16* Qhead = Qh + (size_t)bh * S_LEN * HD;
  const _Float16* Khead = Kh + (size_t)bh * S_LEN * HD;
  const _Float16* Vhead = Vh + (size_t)bh * S_LEN * HD;

  // Q fragments: 16 rows x 64 d -> two A-frags (d 0..31, 32..63)
  v16h qf[2];
  qf[0] = load_frag_a(Qhead + (size_t)q0 * HD, HD, lane);
  qf[1] = load_frag_a(Qhead + (size_t)q0 * HD + 32, HD, lane);

  v8f acc_o[4] = {};
  float mrow[8], lrow[8];
#pragma unroll
  for (int r = 0; r < 8; ++r) { mrow[r] = -1e30f; lrow[r] = 0.f; }

  const float scale = 0.125f;  // 1/sqrt(64)
  _Float16* myP = &sP[wv * 16 * 40];
  const int kv_end = qblk + 128;

  for (int kv0 = 0; kv0 < kv_end; kv0 += 32) {
    __syncthreads();
    {  // cooperative staging: K raw + V transposed
      const int r = tid >> 3, c = (tid & 7) << 3;
      *(uint4*)&sK[r * 72 + c] =
          *(const uint4*)&Khead[(size_t)(kv0 + r) * HD + c];
      v8h vvv = *(const v8h*)&Vhead[(size_t)(kv0 + r) * HD + c];
#pragma unroll
      for (int j = 0; j < 8; ++j) sVT[(c + j) * 40 + r] = vvv[j];
      __builtin_prefetch(&Khead[(size_t)(kv0 + 32 + r) * HD + c], 0, 0);
      __builtin_prefetch(&Vhead[(size_t)(kv0 + 32 + r) * HD + c], 0, 0);
    }
    __syncthreads();

    if (kv0 <= q0 + 15) {  // wave-uniform: skip fully-masked tiles
      // S = Q @ K^T  (16 q-rows x 32 keys)
      v8f sacc[2] = {};
#pragma unroll
      for (int kt = 0; kt < 2; ++kt)
#pragma unroll
        for (int nt = 0; nt < 2; ++nt) {
          v16h bf = load_frag_b(&sK[(nt * 16) * 72 + kt * 32], 72, lane);
          sacc[nt] = wmma16x16x32(qf[kt], bf, sacc[nt]);
        }

      // mask + online softmax (row reductions across 16-lane halves)
      const int col0 = kv0 + (lane & 15);
      const int rowbase = q0 + ((lane >> 4) << 3);
      float pv[2][8], corr[8];
#pragma unroll
      for (int r = 0; r < 8; ++r) {
        const int qpos = rowbase + r;
        float v0 = sacc[0][r] * scale + ((col0      <= qpos) ? 0.f : -1e9f);
        float v1 = sacc[1][r] * scale + ((col0 + 16 <= qpos) ? 0.f : -1e9f);
        float mx = fmaxf(v0, v1);
        mx = fmaxf(mx, XOR1(mx));
        mx = fmaxf(mx, XOR2(mx));
        mx = fmaxf(mx, XOR4(mx));
        mx = fmaxf(mx, XOR8(mx));
        const float mnew = fmaxf(mrow[r], mx);
        corr[r] = __expf(mrow[r] - mnew);
        const float p0 = __expf(v0 - mnew);
        const float p1 = __expf(v1 - mnew);
        float sum = p0 + p1;
        sum += XOR1(sum);
        sum += XOR2(sum);
        sum += XOR4(sum);
        sum += XOR8(sum);
        lrow[r] = lrow[r] * corr[r] + sum;
        mrow[r] = mnew;
        pv[0][r] = p0; pv[1][r] = p1;
      }
#pragma unroll
      for (int nt = 0; nt < 4; ++nt)
#pragma unroll
        for (int r = 0; r < 8; ++r) acc_o[nt][r] *= corr[r];

      // bounce P through per-wave LDS (C-layout -> A-layout); same-wave DS ops
      // are in-order, fence stops compiler reordering.
#pragma unroll
      for (int t = 0; t < 2; ++t)
#pragma unroll
        for (int r = 0; r < 8; ++r)
          myP[(((lane >> 4) << 3) + r) * 40 + t * 16 + (lane & 15)] =
              (_Float16)pv[t][r];
      __asm__ volatile("" ::: "memory");

      v16h pf = load_frag_a(myP, 40, lane);
#pragma unroll
      for (int nt = 0; nt < 4; ++nt) {
        v16h vf = load_frag_b(&sVT[(nt * 16) * 40], 40, lane);
        acc_o[nt] = wmma16x16x32(pf, vf, acc_o[nt]);
      }
    }
  }

  // epilogue: normalize, store context as [B,S,H*64] f16
#pragma unroll
  for (int nt = 0; nt < 4; ++nt)
#pragma unroll
    for (int r = 0; r < 8; ++r) {
      const int s = q0 + ((lane >> 4) << 3) + r;
      const int d = nt * 16 + (lane & 15);
      const float o = acc_o[nt][r] / lrow[r];
      Ch[(size_t)(b * S_LEN + s) * DM + h * HD + d] = (_Float16)o;
    }
}

// --------------------------------------------------------- output projection
__global__ __launch_bounds__(256) void out_gemm(
    const _Float16* __restrict__ Xh, const _Float16* __restrict__ Wh,
    float* __restrict__ out) {
  __shared__ __attribute__((aligned(16))) _Float16 lA[128 * 72];
  __shared__ __attribute__((aligned(16))) _Float16 lB[128 * 72];
  const int tid  = threadIdx.x;
  const int lane = tid & 31, wv = tid >> 5;
  const int wm = (wv & 3) * 32, wn = (wv >> 2) * 64;
  const int bM = blockIdx.x * 128, bN = blockIdx.y * 128;
  const int lr = tid >> 1, lc = (tid & 1) << 5;

  v8f acc[2][4] = {};

  for (int k0 = 0; k0 < DM; k0 += 64) {
    __syncthreads();
    const _Float16* ga = Xh + (size_t)(bM + lr) * DM + k0 + lc;
    const _Float16* gb = Wh + (size_t)(bN + lr) * DM + k0 + lc;
    uint4* sa = (uint4*)&lA[lr * 72 + lc];
    uint4* sb = (uint4*)&lB[lr * 72 + lc];
#pragma unroll
    for (int j = 0; j < 4; ++j) sa[j] = ((const uint4*)ga)[j];
#pragma unroll
    for (int j = 0; j < 4; ++j) sb[j] = ((const uint4*)gb)[j];
    __builtin_prefetch(ga + 64, 0, 0);
    __builtin_prefetch(gb + 64, 0, 0);
    __syncthreads();

#pragma unroll
    for (int kt = 0; kt < 2; ++kt) {
      v16h af[2], bf[4];
#pragma unroll
      for (int mt = 0; mt < 2; ++mt)
        af[mt] = load_frag_a(&lA[(wm + mt * 16) * 72 + kt * 32], 72, lane);
#pragma unroll
      for (int nt = 0; nt < 4; ++nt)
        bf[nt] = load_frag_b(&lB[(wn + nt * 16) * 72 + kt * 32], 72, lane);
#pragma unroll
      for (int mt = 0; mt < 2; ++mt)
#pragma unroll
        for (int nt = 0; nt < 4; ++nt)
          acc[mt][nt] = wmma16x16x32(af[mt], bf[nt], acc[mt][nt]);
    }
  }

#pragma unroll
  for (int mt = 0; mt < 2; ++mt)
#pragma unroll
    for (int nt = 0; nt < 4; ++nt) {
      const int n = bN + wn + nt * 16 + (lane & 15);
#pragma unroll
      for (int r = 0; r < 8; ++r) {
        const int m = bM + wm + mt * 16 + ((lane >> 4) << 3) + r;
        out[(size_t)m * DM + n] = acc[mt][nt][r];
      }
    }
}

// ---------------------------------------------------------------------------
extern "C" void kernel_launch(void* const* d_in, const int* in_sizes, int n_in,
                              void* d_out, int out_size, void* d_ws, size_t ws_size,
                              hipStream_t stream) {
  (void)in_sizes; (void)n_in; (void)out_size; (void)ws_size;
  const float* x    = (const float*)d_in[0];
  const float* cosT = (const float*)d_in[1];
  const float* sinT = (const float*)d_in[2];
  // d_in[3] = causal_mask: recomputed analytically in-kernel
  const float* wq = (const float*)d_in[4];
  const float* wk = (const float*)d_in[5];
  const float* wv = (const float*)d_in[6];
  const float* wo = (const float*)d_in[7];
  float* out = (float*)d_out;

  char* ws = (char*)d_ws;
  const size_t MB = 1024 * 1024;            // workspace layout (88 MiB total)
  _Float16* xh  = (_Float16*)(ws + 0);      // 16 MiB  x as f16
  _Float16* wqh = (_Float16*)(ws + 16 * MB);// 2 MiB each
  _Float16* wkh = (_Float16*)(ws + 18 * MB);
  _Float16* wvh = (_Float16*)(ws + 20 * MB);
  _Float16* woh = (_Float16*)(ws + 22 * MB);
  _Float16* Qh  = (_Float16*)(ws + 24 * MB);// 16 MiB [B,H,S,64]
  _Float16* Kh  = (_Float16*)(ws + 40 * MB);
  _Float16* Vh  = (_Float16*)(ws + 56 * MB);
  _Float16* Chx = (_Float16*)(ws + 72 * MB);// 16 MiB context [B,S,1024]

  const int NX = 4 * S_LEN * DM;
  const int NW = DM * DM;
  cvt_f32_f16<<<NX / 4 / 256, 256, 0, stream>>>(x,  xh,  NX);
  cvt_f32_f16<<<NW / 4 / 256, 256, 0, stream>>>(wq, wqh, NW);
  cvt_f32_f16<<<NW / 4 / 256, 256, 0, stream>>>(wk, wkh, NW);
  cvt_f32_f16<<<NW / 4 / 256, 256, 0, stream>>>(wv, wvh, NW);
  cvt_f32_f16<<<NW / 4 / 256, 256, 0, stream>>>(wo, woh, NW);

  dim3 gGemm(8192 / 128, DM / 128);
  qkv_gemm_rope<<<gGemm, 256, 0, stream>>>(xh, wqh, cosT, sinT, Qh, 1);
  qkv_gemm_rope<<<gGemm, 256, 0, stream>>>(xh, wkh, cosT, sinT, Kh, 1);
  qkv_gemm_rope<<<gGemm, 256, 0, stream>>>(xh, wvh, cosT, sinT, Vh, 0);

  dim3 gAttn(4 * NH, S_LEN / 128);
  flash_attn<<<gAttn, 256, 0, stream>>>(Qh, Kh, Vh, Chx);

  out_gemm<<<gGemm, 256, 0, stream>>>(Chx, woh, out);
}
